// SyntaxAwareDecoder_21225728376938
// MI455X (gfx1250) — matrix operand order repaired and checked by reading, not compiled, not run on hardware
//
#include <hip/hip_runtime.h>
#include <hip/hip_bf16.h>
#include <cstdint>

#define DM 2048   // d_model
#define SQ 8192   // encoder sequence
#define VV 32002  // vocab + 2
#define RR 7      // num_rel

typedef __attribute__((ext_vector_type(16))) __bf16 v16bf;
typedef __attribute__((ext_vector_type(2)))  __bf16 v2bf;
typedef __attribute__((ext_vector_type(8)))  float  v8f;
typedef int b128_t __attribute__((vector_size(16)));   // type expected by async-LDS builtin

#if __has_builtin(__builtin_amdgcn_global_load_async_to_lds_b128) && \
    __has_builtin(__builtin_amdgcn_s_wait_asynccnt)
#define USE_ASYNC_LDS 1
#endif

// ---------- helpers ----------
__device__ __forceinline__ unsigned short f2bf(float f) {
  unsigned u = __float_as_uint(f);
  unsigned r = (u + 0x7FFFu + ((u >> 16) & 1u)) >> 16;   // round-to-nearest-even
  return (unsigned short)r;
}
__device__ __forceinline__ float sigmoidf(float x) { return 1.0f / (1.0f + __expf(-x)); }

// Stage 16 bf16 (two 16B pieces) from global into LDS.
// CDNA5 async global->LDS path when available (ASYNCcnt), else VGPR round-trip.
__device__ __forceinline__ void stage_b(const unsigned short* __restrict__ src,
                                        unsigned short* dst) {
#ifdef USE_ASYNC_LDS
  __builtin_amdgcn_global_load_async_to_lds_b128((b128_t*)src, (b128_t*)dst, 0, 0);
  __builtin_amdgcn_global_load_async_to_lds_b128((b128_t*)(src + 8), (b128_t*)(dst + 8), 0, 0);
#else
  uint4 b0 = *(const uint4*)(src);
  uint4 b1 = *(const uint4*)(src + 8);
  *(uint4*)(dst) = b0;
  *(uint4*)(dst + 8) = b1;
#endif
}
__device__ __forceinline__ void stage_wait() {
#ifdef USE_ASYNC_LDS
  __builtin_amdgcn_s_wait_asynccnt(0);
#endif
}

// ---------- fp32 -> bf16 conversion (W_enc, 8 MB bf16, stays L2-resident) ----------
__global__ void cvt_bf16_kernel(const float* __restrict__ w,
                                unsigned short* __restrict__ out, int n4) {
  int i = blockIdx.x * 256 + threadIdx.x;
  if (i >= n4) return;
  float4 f = ((const float4*)w)[i];
  uint2 p;
  p.x = (unsigned)f2bf(f.x) | ((unsigned)f2bf(f.y) << 16);
  p.y = (unsigned)f2bf(f.z) | ((unsigned)f2bf(f.w) << 16);
  ((uint2*)out)[i] = p;
}

// ---------- GRU gate matvecs: gi[r]=Wih[r].x+bih[r], gh[r]=Whh[r].h+bhh[r] ----------
// one wave per row; 3D=6144 rows -> grid 768 x 256thr
__global__ void gru_gates_kernel(const float* __restrict__ Wih, const float* __restrict__ Whh,
                                 const float* __restrict__ bih, const float* __restrict__ bhh,
                                 const float* __restrict__ x,   const float* __restrict__ h,
                                 float* __restrict__ gi, float* __restrict__ gh) {
  int row  = blockIdx.x * 8 + (threadIdx.x >> 5);
  int lane = threadIdx.x & 31;
  const float* wi = Wih + (size_t)row * DM;
  const float* wh = Whh + (size_t)row * DM;
  float a1 = 0.f, a2 = 0.f;
  for (int k = lane * 4; k < DM; k += 128) {
    float4 xv = *(const float4*)(x + k);
    float4 hv = *(const float4*)(h + k);
    float4 w1 = *(const float4*)(wi + k);
    float4 w2 = *(const float4*)(wh + k);
    a1 += w1.x * xv.x + w1.y * xv.y + w1.z * xv.z + w1.w * xv.w;
    a2 += w2.x * hv.x + w2.y * hv.y + w2.z * hv.z + w2.w * hv.w;
  }
  for (int off = 16; off > 0; off >>= 1) {
    a1 += __shfl_xor(a1, off, 32);
    a2 += __shfl_xor(a2, off, 32);
  }
  if (lane == 0) { gi[row] = a1 + bih[row]; gh[row] = a2 + bhh[row]; }
}

// ---------- GRU combine: r,z,n gates -> h' ----------
__global__ void gru_combine_kernel(const float* __restrict__ gi, const float* __restrict__ gh,
                                   const float* __restrict__ h, float* __restrict__ out) {
  int d = blockIdx.x * 256 + threadIdx.x;
  float r = sigmoidf(gi[d] + gh[d]);
  float z = sigmoidf(gi[DM + d] + gh[DM + d]);
  float n = tanhf(gi[2 * DM + d] + r * gh[2 * DM + d]);
  out[d] = (1.0f - z) * n + z * h[d];
}

// ---------- pre[j] = b_enc[j] + (W_co c_o + b_co)[j] + (W_cov cov + b_cov)[j] ----------
__global__ void attn_pre_kernel(const float* __restrict__ W_co,  const float* __restrict__ b_co,
                                const float* __restrict__ W_cov, const float* __restrict__ b_cov,
                                const float* __restrict__ b_enc, const float* __restrict__ c_o,
                                const float* __restrict__ cov_vec, float* __restrict__ pre) {
  int j    = blockIdx.x * 8 + (threadIdx.x >> 5);
  int lane = threadIdx.x & 31;
  const float* w1 = W_co  + (size_t)j * DM;
  const float* w2 = W_cov + (size_t)j * DM;
  float a1 = 0.f, a2 = 0.f;
  for (int k = lane * 4; k < DM; k += 128) {
    float4 cv = *(const float4*)(c_o + k);
    float4 vv = *(const float4*)(cov_vec + k);
    float4 u1 = *(const float4*)(w1 + k);
    float4 u2 = *(const float4*)(w2 + k);
    a1 += u1.x * cv.x + u1.y * cv.y + u1.z * cv.z + u1.w * cv.w;
    a2 += u2.x * vv.x + u2.y * vv.y + u2.z * vv.z + u2.w * vv.w;
  }
  for (int off = 16; off > 0; off >>= 1) {
    a1 += __shfl_xor(a1, off, 32);
    a2 += __shfl_xor(a2, off, 32);
  }
  if (lane == 0) pre[j] = a1 + a2 + b_co[j] + b_cov[j] + b_enc[j];
}

__global__ void scores_init_kernel(const float* __restrict__ b_score, float* __restrict__ scores) {
  scores[blockIdx.x * 256 + threadIdx.x] = b_score[0];
}

// ---------- fused WMMA GEMM + tanh + score reduction ----------
// scores[s] += sum_j tanh( (enc @ W_enc^T)[s,j] + pre[j] ) * wscore[j]
// Block = 8 waves, M-block 128 rows, N-block 128 cols; K in chunks of 32.
// B chunk (128 cols x 32 K, bf16) double-buffered in LDS, shared by all waves,
// staged with async global->LDS copies where available (one barrier per K-step).
__global__ __launch_bounds__(256) void gemm_scores_kernel(
    const float* __restrict__ enc, const unsigned short* __restrict__ Wb,
    const float* __restrict__ pre, const float* __restrict__ wscore,
    float* __restrict__ scores) {
  // per buffer: [khalf][col][16 K], col stride padded to 24 ushorts (48 B, 16B-aligned)
  __shared__ unsigned short ldsb[2][2 * 128 * 24];

  const int tid  = threadIdx.x;
  const int wave = tid >> 5;
  const int lane = tid & 31;
  const int n    = lane & 15;     // A row-in-strip / B column / D column
  const int kh   = lane >> 4;     // K half selector per ISA fragment layout
  const int Mbase = blockIdx.x * 128;
  const int Nbase = blockIdx.y * 128;

  const float* arow = enc + (size_t)(Mbase + wave * 16 + n) * DM;

  // B-loader indices: 2 threads per column, 16 bf16 (32 B) each
  const int lc = tid >> 1;
  const int lh = tid & 1;
  const unsigned short* bsrc = Wb + (size_t)(Nbase + lc) * DM + lh * 16;
  const int bdoff = lh * 128 * 24 + lc * 24;   // ushort offset within one buffer

  v8f acc[8];
#pragma unroll
  for (int j = 0; j < 8; ++j) acc[j] = v8f{0.f, 0.f, 0.f, 0.f, 0.f, 0.f, 0.f, 0.f};

  // prologue: stage K-chunk 0 into buffer 0
  stage_b(bsrc, &ldsb[0][bdoff]);
  stage_wait();
  __syncthreads();

  for (int kb = 0; kb < DM; kb += 32) {
    const int cur = (kb >> 5) & 1;
    // prefetch next K-chunk into the other buffer while computing this one
    if (kb + 32 < DM) stage_b(bsrc + kb + 32, &ldsb[cur ^ 1][bdoff]);

    // A fragment: lane holds A[m, kb + kh*8 + 0..7] and A[m, kb + 16 + kh*8 + 0..7]
    float4 f0 = *(const float4*)(arow + kb + kh * 8);
    float4 f1 = *(const float4*)(arow + kb + kh * 8 + 4);
    float4 f2 = *(const float4*)(arow + kb + 16 + kh * 8);
    float4 f3 = *(const float4*)(arow + kb + 16 + kh * 8 + 4);
#if __has_builtin(__builtin_amdgcn_cvt_pk_bf16_f32)
    union { v16bf v; v2bf p[8]; } A;
    A.p[0] = __builtin_amdgcn_cvt_pk_bf16_f32(f0.x, f0.y);
    A.p[1] = __builtin_amdgcn_cvt_pk_bf16_f32(f0.z, f0.w);
    A.p[2] = __builtin_amdgcn_cvt_pk_bf16_f32(f1.x, f1.y);
    A.p[3] = __builtin_amdgcn_cvt_pk_bf16_f32(f1.z, f1.w);
    A.p[4] = __builtin_amdgcn_cvt_pk_bf16_f32(f2.x, f2.y);
    A.p[5] = __builtin_amdgcn_cvt_pk_bf16_f32(f2.z, f2.w);
    A.p[6] = __builtin_amdgcn_cvt_pk_bf16_f32(f3.x, f3.y);
    A.p[7] = __builtin_amdgcn_cvt_pk_bf16_f32(f3.z, f3.w);
#else
    union { v16bf v; unsigned short s[16]; } A;
    A.s[0]  = f2bf(f0.x); A.s[1]  = f2bf(f0.y); A.s[2]  = f2bf(f0.z); A.s[3]  = f2bf(f0.w);
    A.s[4]  = f2bf(f1.x); A.s[5]  = f2bf(f1.y); A.s[6]  = f2bf(f1.z); A.s[7]  = f2bf(f1.w);
    A.s[8]  = f2bf(f2.x); A.s[9]  = f2bf(f2.y); A.s[10] = f2bf(f2.z); A.s[11] = f2bf(f2.w);
    A.s[12] = f2bf(f3.x); A.s[13] = f2bf(f3.y); A.s[14] = f2bf(f3.z); A.s[15] = f2bf(f3.w);
#endif

#pragma unroll
    for (int jt = 0; jt < 8; ++jt) {
      const unsigned short* bp = &ldsb[cur][(jt * 16 + n) * 24 + kh * 8];
      union { v16bf v; uint4 u[2]; } B;
      B.u[0] = *(const uint4*)(bp);
      B.u[1] = *(const uint4*)(bp + 128 * 24);
      acc[jt] = __builtin_amdgcn_wmma_f32_16x16x32_bf16(
          false, A.v, false, B.v, (short)0, acc[jt], false, false);
    }

    stage_wait();       // our async stores into the next buffer are done
    __syncthreads();    // everyone finished reading cur & staging cur^1
  }

  // epilogue: tanh + dot with wscore, then cross-lane row reduction
  float sacc[8] = {0.f, 0.f, 0.f, 0.f, 0.f, 0.f, 0.f, 0.f};
#pragma unroll
  for (int jt = 0; jt < 8; ++jt) {
    int col = Nbase + jt * 16 + n;
    float p = pre[col];
    float w = wscore[col];
#pragma unroll
    for (int i = 0; i < 8; ++i) sacc[i] += tanhf(acc[jt][i] + p) * w;
  }
#pragma unroll
  for (int i = 0; i < 8; ++i) {
    float v = sacc[i];
    for (int off = 8; off > 0; off >>= 1) v += __shfl_xor(v, off, 16);
    if (n == 0) atomicAdd(&scores[Mbase + wave * 16 + i + 8 * kh], v);
  }
}

// ---------- softmax over S=8192, single workgroup ----------
__global__ void softmax_kernel(const float* __restrict__ scores, float* __restrict__ alpha) {
  __shared__ float red[1024];
  int tid = threadIdx.x;
  float s[8];
  float mx = -3.4e38f;
#pragma unroll
  for (int i = 0; i < 8; ++i) { s[i] = scores[tid + i * 1024]; mx = fmaxf(mx, s[i]); }
  red[tid] = mx; __syncthreads();
  for (int off = 512; off > 0; off >>= 1) {
    if (tid < off) red[tid] = fmaxf(red[tid], red[tid + off]);
    __syncthreads();
  }
  mx = red[0];
  __syncthreads();
  float e[8], sum = 0.f;
#pragma unroll
  for (int i = 0; i < 8; ++i) { e[i] = expf(s[i] - mx); sum += e[i]; }
  red[tid] = sum; __syncthreads();
  for (int off = 512; off > 0; off >>= 1) {
    if (tid < off) red[tid] += red[tid + off];
    __syncthreads();
  }
  float inv = 1.0f / red[0];
#pragma unroll
  for (int i = 0; i < 8; ++i) alpha[tid + i * 1024] = e[i] * inv;
}

// ---------- ctx[d] = sum_s alpha[s] * enc[s,d] ----------
__global__ void ctx_kernel(const float* __restrict__ alpha, const float* __restrict__ enc,
                           float* __restrict__ ctx) {
  __shared__ float la[256];
  int d = blockIdx.x * 256 + threadIdx.x;
  float acc = 0.f;
  for (int s0 = 0; s0 < SQ; s0 += 256) {
    __syncthreads();
    la[threadIdx.x] = alpha[s0 + threadIdx.x];
    __syncthreads();
    for (int t = 0; t < 256; ++t) acc += la[t] * enc[(size_t)(s0 + t) * DM + d];
  }
  ctx[d] = acc;
}

// ---------- logits head: out[r] = W[r].x + b[r], wave per row ----------
__global__ void head_kernel(const float* __restrict__ W, const float* __restrict__ b,
                            const float* __restrict__ x, float* __restrict__ out, int nrows) {
  int row  = blockIdx.x * 8 + (threadIdx.x >> 5);
  if (row >= nrows) return;
  int lane = threadIdx.x & 31;
  const float* wr = W + (size_t)row * DM;
  float a = 0.f;
  for (int k = lane * 4; k < DM; k += 128) {
    float4 xv = *(const float4*)(x + k);
    float4 wv = *(const float4*)(wr + k);
    a += wv.x * xv.x + wv.y * xv.y + wv.z * xv.z + wv.w * xv.w;
  }
  for (int off = 16; off > 0; off >>= 1) a += __shfl_xor(a, off, 32);
  if (lane == 0) out[row] = a + b[row];
}

// ---------- launch ----------
extern "C" void kernel_launch(void* const* d_in, const int* in_sizes, int n_in,
                              void* d_out, int out_size, void* d_ws, size_t ws_size,
                              hipStream_t stream) {
  const float* enc      = (const float*)d_in[0];
  const float* c_prev   = (const float*)d_in[1];
  const float* partner  = (const float*)d_in[2];
  const float* cov_vec  = (const float*)d_in[3];
  const float* Wih_a    = (const float*)d_in[4];
  const float* Whh_a    = (const float*)d_in[5];
  const float* bih_a    = (const float*)d_in[6];
  const float* bhh_a    = (const float*)d_in[7];
  const float* Wih_b    = (const float*)d_in[8];
  const float* Whh_b    = (const float*)d_in[9];
  const float* bih_b    = (const float*)d_in[10];
  const float* bhh_b    = (const float*)d_in[11];
  const float* W_enc    = (const float*)d_in[12];
  const float* b_enc    = (const float*)d_in[13];
  const float* W_co     = (const float*)d_in[14];
  const float* b_co     = (const float*)d_in[15];
  const float* W_cov    = (const float*)d_in[16];
  const float* b_cov    = (const float*)d_in[17];
  const float* W_score  = (const float*)d_in[18];  // [1, D]
  const float* b_score  = (const float*)d_in[19];
  const float* W_sym    = (const float*)d_in[20];
  const float* b_sym    = (const float*)d_in[21];
  const float* W_rel    = (const float*)d_in[22];
  const float* b_rel    = (const float*)d_in[23];

  // workspace layout (bytes)
  char* ws = (char*)d_ws;
  unsigned short* Wb = (unsigned short*)(ws);                 // D*D bf16 = 8,388,608 B
  float* gi_a   = (float*)(ws + 8388608);                     // 3D f32
  float* gh_a   = (float*)(ws + 8388608 + 24576);
  float* c_o    = (float*)(ws + 8388608 + 49152);             // D f32
  float* pre    = (float*)(ws + 8388608 + 57344);             // D f32
  float* scores = (float*)(ws + 8388608 + 65536);             // S f32
  float* ctx    = (float*)(ws + 8388608 + 98304);             // D f32
  float* gi_b   = (float*)(ws + 8388608 + 106496);            // 3D f32
  float* gh_b   = (float*)(ws + 8388608 + 131072);

  // output segments: [sym V][rel R][c_beta D][alpha S]
  float* out    = (float*)d_out;
  float* sym    = out;
  float* rel    = out + VV;
  float* c_beta = out + VV + RR;
  float* alpha  = out + VV + RR + DM;

  // 1) W_enc -> bf16 (L2-resident, reused by every GEMM block)
  cvt_bf16_kernel<<<(DM * DM / 4 + 255) / 256, 256, 0, stream>>>(W_enc, Wb, DM * DM / 4);
  // 2) GRU alpha
  gru_gates_kernel<<<(3 * DM) / 8, 256, 0, stream>>>(Wih_a, Whh_a, bih_a, bhh_a,
                                                     partner, c_prev, gi_a, gh_a);
  gru_combine_kernel<<<DM / 256, 256, 0, stream>>>(gi_a, gh_a, c_prev, c_o);
  // 3) pre-vector + score bias init
  attn_pre_kernel<<<DM / 8, 256, 0, stream>>>(W_co, b_co, W_cov, b_cov, b_enc,
                                              c_o, cov_vec, pre);
  scores_init_kernel<<<SQ / 256, 256, 0, stream>>>(b_score, scores);
  // 4) fused WMMA attention-score GEMM
  gemm_scores_kernel<<<dim3(SQ / 128, DM / 128), 256, 0, stream>>>(enc, Wb, pre,
                                                                   W_score, scores);
  // 5) softmax -> alpha (in d_out)
  softmax_kernel<<<1, 1024, 0, stream>>>(scores, alpha);
  // 6) context vector
  ctx_kernel<<<DM / 256, 256, 0, stream>>>(alpha, enc, ctx);
  // 7) GRU beta
  gru_gates_kernel<<<(3 * DM) / 8, 256, 0, stream>>>(Wih_b, Whh_b, bih_b, bhh_b,
                                                     ctx, c_o, gi_b, gh_b);
  gru_combine_kernel<<<DM / 256, 256, 0, stream>>>(gi_b, gh_b, c_o, c_beta);
  // 8) heads
  head_kernel<<<(VV + 7) / 8, 256, 0, stream>>>(W_sym, b_sym, c_beta, sym, VV);
  head_kernel<<<1, 256, 0, stream>>>(W_rel, b_rel, c_beta, rel, RR);
}
